// GroupedQueryAttention_37237366456736
// MI455X (gfx1250) — matrix operand order repaired and checked by reading, not compiled, not run on hardware
//
#include <hip/hip_runtime.h>
#include <hip/hip_bf16.h>

// ---------------------------------------------------------------- constants
constexpr int BATCH = 2;
constexpr int SEQ   = 2048;
constexpr int DMODEL= 2048;
constexpr int NH    = 32;
constexpr int NKV   = 8;
constexpr int DHEAD = 64;
constexpr float NEGM = -1e9f;

typedef __attribute__((ext_vector_type(16))) __bf16 v16bf;
typedef __attribute__((ext_vector_type(8)))  float  v8f;
typedef __attribute__((ext_vector_type(4)))  unsigned int u32x4;
typedef __attribute__((ext_vector_type(8)))  int i32x8;
typedef __attribute__((ext_vector_type(4)))  int i32x4;

union Frag16 { v16bf v; uint4 q[2]; unsigned short h[16]; };

static __device__ __forceinline__ __bf16 f2bf(float f) {
  unsigned int u = __builtin_bit_cast(unsigned int, f);
  unsigned int r = (u + 0x7FFFu + ((u >> 16) & 1u)) >> 16;
  unsigned short s = (unsigned short)r;
  return __builtin_bit_cast(__bf16, s);
}
static __device__ __forceinline__ float bf2f(__bf16 b) {
  unsigned int u = ((unsigned int)__builtin_bit_cast(unsigned short, b)) << 16;
  return __builtin_bit_cast(float, u);
}

// ---------------------------------------------------------------- TDM (Tensor Data Mover)
#if defined(__has_builtin)
#  if __has_builtin(__builtin_amdgcn_tensor_load_to_lds)
#    define HAVE_TDM 1
#  endif
#endif
#ifndef HAVE_TDM
#  define HAVE_TDM 0
#endif

#if HAVE_TDM
// 2D bf16 tile DMA: global[tile_y rows x tile_x elems, row stride stride_x elems] -> LDS
// (contiguous row-major). D# layout per CDNA5 ISA 8.3/8.4: count=1, type=2, data_size=1 (2B).
static __device__ __forceinline__ void tdm_load_2d(unsigned lds_off, const void* gaddr,
                                                   unsigned tile_x, unsigned tile_y,
                                                   unsigned tensor_x, unsigned tensor_y,
                                                   unsigned stride_x) {
  unsigned long long ga = (unsigned long long)(size_t)gaddr;
  u32x4 g0;
  g0[0] = 1u;                                            // count=1, user descriptor
  g0[1] = lds_off;                                       // lds_addr (bytes)
  g0[2] = (unsigned)(ga & 0xFFFFFFFFu);                  // global_addr[31:0]
  g0[3] = (unsigned)((ga >> 32) & 0x01FFFFFFu) | (2u << 30); // addr[56:32] | type=2
  i32x8 g1;
  g1[0] = (int)(1u << 16);                               // wg_mask=0, data_size=1 (2B)
  g1[1] = (int)((tensor_x & 0xFFFFu) << 16);             // tensor_dim0[15:0] @ bit48
  g1[2] = (int)(((tensor_x >> 16) & 0xFFFFu) | ((tensor_y & 0xFFFFu) << 16));
  g1[3] = (int)(((tensor_y >> 16) & 0xFFFFu) | ((tile_x & 0xFFFFu) << 16)); // tile_dim0
  g1[4] = (int)(tile_y & 0xFFFFu);                       // tile_dim1 (tile_dim2=0)
  g1[5] = (int)stride_x;                                 // tensor_dim0_stride[31:0]
  g1[6] = 0;                                             // stride[47:32]=0, dim1_stride=0
  g1[7] = 0;
  i32x4 z4 = {0, 0, 0, 0};
#if __clang_major__ >= 23
  i32x8 z8 = {0, 0, 0, 0, 0, 0, 0, 0};
  __builtin_amdgcn_tensor_load_to_lds(g0, g1, z4, z4, z8, 0);
#else
  __builtin_amdgcn_tensor_load_to_lds(g0, g1, z4, z4, 0);
#endif
}
static __device__ __forceinline__ unsigned lds_off_of(const void* p) {
  return (unsigned)(size_t)p; // low 32 bits of flat LDS address = LDS byte offset
}
#endif

// ---------------------------------------------------------------- fp32 -> bf16
__global__ void k_cvt(const float* __restrict__ in, __bf16* __restrict__ out, int n) {
  for (int i = blockIdx.x * blockDim.x + threadIdx.x; i < n; i += gridDim.x * blockDim.x)
    out[i] = f2bf(in[i]);
}

// ---------------------------------------------------------------- RoPE tables [SEQ][DHEAD]
__global__ void k_rope_tab(float* __restrict__ cosT, float* __restrict__ sinT) {
  int i = blockIdx.x * blockDim.x + threadIdx.x;
  if (i >= SEQ * DHEAD) return;
  int l = i / DHEAD, d = i % DHEAD;
  int half = d & 31; // emb = concat(freqs, freqs)
  float invf = __powf(10000.0f, -(2.0f * (float)half) / (float)DHEAD);
  float fr = (float)l * invf;
  cosT[i] = __cosf(fr);
  sinT[i] = __sinf(fr);
}

// ---------------------------------------------------------------- tiled WMMA GEMM
// C[M,N] = A[M,K] (bf16, row-major) * Bw[N,K]^T (bf16, row = output feature)
// 256 threads = 8 waves, block tile 128x64, wave tile 32x32 = 2x2 WMMA frags.
// TDM path: double-buffered LDS; wave 0 issues tile DMA for chunk i+1 while all
// waves run WMMA on chunk i (s_wait_tensorcnt 2 drains only the older two transfers).
template <bool F32OUT>
__global__ void __launch_bounds__(256, 1)
k_gemm(const __bf16* __restrict__ A, const __bf16* __restrict__ Bw,
       void* __restrict__ Cout, int M, int N, int K) {
  constexpr int BM = 128, BN = 64, BK = 32;
#if HAVE_TDM
  __shared__ __align__(32) __bf16 sA[2][BM][BK]; // 16 KB
  __shared__ __align__(32) __bf16 sB[2][BN][BK]; //  8 KB
#else
  __shared__ __align__(32) __bf16 sA[1][BM][BK];
  __shared__ __align__(32) __bf16 sB[1][BN][BK];
#endif

  const int tid = threadIdx.x, wave = tid >> 5, lane = tid & 31;
  const int l16 = lane & 15, halfl = lane >> 4;
  const int m0 = blockIdx.y * BM, n0 = blockIdx.x * BN;
  const int wm = (wave >> 1) * 32, wn = (wave & 1) * 32;

  v8f acc[2][2] = {};
  const int nit = K / BK;

#if HAVE_TDM
  if (wave == 0) { // prologue: DMA chunk 0 into buffer 0
    tdm_load_2d(lds_off_of(&sA[0][0][0]), A + (size_t)m0 * K, BK, BM, (unsigned)K,
                (unsigned)M, (unsigned)K);
    tdm_load_2d(lds_off_of(&sB[0][0][0]), Bw + (size_t)n0 * K, BK, BN, (unsigned)K,
                (unsigned)N, (unsigned)K);
  }
#endif

  for (int it = 0; it < nit; ++it) {
    const int k0 = it * BK;
#if HAVE_TDM
    const int cur = it & 1, nxt = cur ^ 1;
    if (wave == 0) {
      if (it + 1 < nit) { // overlap: issue next chunk's DMA, wait only for current
        tdm_load_2d(lds_off_of(&sA[nxt][0][0]), A + (size_t)m0 * K + k0 + BK, BK, BM,
                    (unsigned)K, (unsigned)M, (unsigned)K);
        tdm_load_2d(lds_off_of(&sB[nxt][0][0]), Bw + (size_t)n0 * K + k0 + BK, BK, BN,
                    (unsigned)K, (unsigned)N, (unsigned)K);
        __builtin_amdgcn_s_wait_tensorcnt(2); // <=2 outstanding: current chunk landed
      } else {
        __builtin_amdgcn_s_wait_tensorcnt(0);
      }
    }
#else
    const int cur = 0;
    { // manual staging fallback
      const int arow = tid >> 1, akc = (tid & 1) * 16;
      const int brow = tid >> 2, bkc = (tid & 3) * 8;
      const uint4* sa = (const uint4*)(A + (size_t)(m0 + arow) * K + k0 + akc);
      *(uint4*)&sA[0][arow][akc]     = sa[0];
      *(uint4*)&sA[0][arow][akc + 8] = sa[1];
      *(uint4*)&sB[0][brow][bkc] = *(const uint4*)(Bw + (size_t)(n0 + brow) * K + k0 + bkc);
      if (k0 + BK < K) {
        __builtin_prefetch(A + (size_t)(m0 + arow) * K + k0 + BK + akc, 0, 1);
        __builtin_prefetch(Bw + (size_t)(n0 + brow) * K + k0 + BK + bkc, 0, 1);
      }
    }
#endif
    __syncthreads(); // current chunk visible to all waves

    v16bf af[2], bfr[2];
#pragma unroll
    for (int fm = 0; fm < 2; ++fm) { // A frag 16x32: lane row = l16; K = halfl*8..+7, +16
      Frag16 f;
      f.q[0] = *(const uint4*)&sA[cur][wm + 16 * fm + l16][halfl * 8];
      f.q[1] = *(const uint4*)&sA[cur][wm + 16 * fm + l16][halfl * 8 + 16];
      af[fm] = f.v;
    }
#pragma unroll
    for (int fn = 0; fn < 2; ++fn) { // B frag 32x16: lane n = l16; K = halfl*8..+7, +16
      Frag16 f;
      f.q[0] = *(const uint4*)&sB[cur][wn + 16 * fn + l16][halfl * 8];
      f.q[1] = *(const uint4*)&sB[cur][wn + 16 * fn + l16][halfl * 8 + 16];
      bfr[fn] = f.v;
    }
#pragma unroll
    for (int fm = 0; fm < 2; ++fm)
#pragma unroll
      for (int fn = 0; fn < 2; ++fn)
        acc[fm][fn] = __builtin_amdgcn_wmma_f32_16x16x32_bf16(
            false, af[fm], false, bfr[fn], (short)0, acc[fm][fn], false, false);
    __syncthreads(); // all reads done before buffer is overwritten
  }

#pragma unroll
  for (int fm = 0; fm < 2; ++fm)
#pragma unroll
    for (int fn = 0; fn < 2; ++fn)
#pragma unroll
      for (int i = 0; i < 8; ++i) {
        int gm = m0 + wm + 16 * fm + i + 8 * halfl;
        int gn = n0 + wn + 16 * fn + l16;
        if (F32OUT) ((float*)Cout)[(size_t)gm * N + gn] = acc[fm][fn][i];
        else        ((__bf16*)Cout)[(size_t)gm * N + gn] = f2bf(acc[fm][fn][i]);
      }
}

// ---------------------------------------------------------------- RoPE + head permute
// src [BATCH, SEQ, heads*DHEAD] -> dst [BATCH, heads, SEQ, DHEAD]
__global__ void k_rope_perm(const __bf16* __restrict__ src, __bf16* __restrict__ dst,
                            const float* __restrict__ cosT, const float* __restrict__ sinT,
                            int heads, int dorope) {
  int total = BATCH * SEQ * heads * DHEAD;
  for (int idx = blockIdx.x * blockDim.x + threadIdx.x; idx < total;
       idx += gridDim.x * blockDim.x) {
    int d = idx % DHEAD;
    int h = (idx / DHEAD) % heads;
    int l = (idx / (DHEAD * heads)) % SEQ;
    int b = idx / (DHEAD * heads * SEQ);
    const __bf16* row = src + ((size_t)(b * SEQ + l) * heads + h) * DHEAD;
    float v = bf2f(row[d]);
    float o = v;
    if (dorope) {
      float c = cosT[l * DHEAD + d], s = sinT[l * DHEAD + d];
      float r = (d < DHEAD / 2) ? -bf2f(row[d + DHEAD / 2]) : bf2f(row[d - DHEAD / 2]);
      o = v * c + r * s;
    }
    dst[(((size_t)b * heads + h) * SEQ + l) * DHEAD + d] = f2bf(o);
  }
}

// ---------------------------------------------------------------- flash attention
// grid (SEQ/64, NH, BATCH); block 128 = 4 waves, each wave owns 16 q rows.
__global__ void __launch_bounds__(128, 1)
k_flash(const __bf16* __restrict__ Qh, const __bf16* __restrict__ Kh,
        const __bf16* __restrict__ Vh, __bf16* __restrict__ Oout) {
  constexpr int KT = 32;
  __shared__ __align__(32) __bf16 sK[KT][DHEAD];   // [key][hd], TDM target
  __shared__ __align__(32) __bf16 sVt[DHEAD][KT];  // transposed [hd][key]
  __shared__ __align__(32) __bf16 sP[4][16][KT];   // per-wave P

  const int qt = blockIdx.x, h = blockIdx.y, b = blockIdx.z;
  const int hk = h / (NH / NKV);
  const int tid = threadIdx.x, wave = tid >> 5, lane = tid & 31;
  const int l16 = lane & 15, halfl = lane >> 4;

  const __bf16* Qbase = Qh + (((size_t)b * NH + h) * SEQ + qt * 64 + wave * 16) * DHEAD;
  const __bf16* Kbase = Kh + ((size_t)b * NKV + hk) * SEQ * DHEAD;
  const __bf16* Vbase = Vh + ((size_t)b * NKV + hk) * SEQ * DHEAD;

  v16bf qf[2];
#pragma unroll
  for (int c = 0; c < 2; ++c) { // Q frag (A layout), two hd-chunks
    const __bf16* r = Qbase + (size_t)l16 * DHEAD + c * 32 + halfl * 8;
    Frag16 f;
    f.q[0] = *(const uint4*)r;
    f.q[1] = *(const uint4*)(r + 16);
    qf[c] = f.v;
  }

  v8f o[4] = {};
  float mrow[8], lrow[8];
#pragma unroll
  for (int i = 0; i < 8; ++i) { mrow[i] = -3e38f; lrow[i] = 0.f; }

  const int srow = tid >> 2, sseg = (tid & 3) * 16;
  const int ntiles = (qt * 64 + 64) / KT;
  const float scale = 0.125f; // 1/sqrt(64)

  for (int t = 0; t < ntiles; ++t) {
    const int kbase = t * KT;
#if HAVE_TDM
    if (wave == 0) // K tile via TDM (32 keys x 64 hd, row-major)
      tdm_load_2d(lds_off_of(&sK[0][0]), Kbase + (size_t)kbase * DHEAD, DHEAD, KT,
                  DHEAD, SEQ, DHEAD);
#else
    {
      const uint4* ks = (const uint4*)(Kbase + (size_t)(kbase + srow) * DHEAD + sseg);
      *(uint4*)&sK[srow][sseg]     = ks[0];
      *(uint4*)&sK[srow][sseg + 8] = ks[1];
    }
#endif
    { // V tile staged transposed (TDM cannot transpose)
      const __bf16* vs = Vbase + (size_t)(kbase + srow) * DHEAD + sseg;
      Frag16 tv;
      tv.q[0] = *(const uint4*)vs;
      tv.q[1] = *(const uint4*)(vs + 8);
#pragma unroll
      for (int j = 0; j < 16; ++j) sVt[sseg + j][srow] = tv.h[j];
    }
#if HAVE_TDM
    if (wave == 0) __builtin_amdgcn_s_wait_tensorcnt(0);
#endif
    __syncthreads();

    // S = Q K^T : two 16-key groups, contraction over DHEAD in 2 WMMAs each
    v8f s[2] = {};
#pragma unroll
    for (int g = 0; g < 2; ++g)
#pragma unroll
      for (int c = 0; c < 2; ++c) {
        Frag16 f; // B frag: n = key(16g+l16), kk = hd c*32 + halfl*8..+7, +16
        f.q[0] = *(const uint4*)&sK[16 * g + l16][c * 32 + halfl * 8];
        f.q[1] = *(const uint4*)&sK[16 * g + l16][c * 32 + halfl * 8 + 16];
        s[g] = __builtin_amdgcn_wmma_f32_16x16x32_bf16(false, qf[c], false, f.v,
                                                       (short)0, s[g], false, false);
      }

    // online softmax over this tile's 32 keys
    float alpha[8];
#pragma unroll
    for (int i = 0; i < 8; ++i) {
      int qglob = qt * 64 + wave * 16 + 8 * halfl + i;
      int key0 = kbase + l16, key1 = kbase + 16 + l16;
      float v0 = s[0][i] * scale + ((key0 <= qglob) ? 0.f : NEGM);
      float v1 = s[1][i] * scale + ((key1 <= qglob) ? 0.f : NEGM);
      float mx = fmaxf(v0, v1);
      mx = fmaxf(mx, __shfl_xor(mx, 1, 32));
      mx = fmaxf(mx, __shfl_xor(mx, 2, 32));
      mx = fmaxf(mx, __shfl_xor(mx, 4, 32));
      mx = fmaxf(mx, __shfl_xor(mx, 8, 32));
      float mnew = fmaxf(mrow[i], mx);
      alpha[i] = __expf(mrow[i] - mnew);
      mrow[i] = mnew;
      v0 = __expf(v0 - mnew);
      v1 = __expf(v1 - mnew);
      float rs = v0 + v1;
      rs += __shfl_xor(rs, 1, 32);
      rs += __shfl_xor(rs, 2, 32);
      rs += __shfl_xor(rs, 4, 32);
      rs += __shfl_xor(rs, 8, 32);
      lrow[i] = lrow[i] * alpha[i] + rs;
      s[0][i] = v0; s[1][i] = v1;
    }
#pragma unroll
    for (int fn = 0; fn < 4; ++fn)
#pragma unroll
      for (int i = 0; i < 8; ++i) o[fn][i] *= alpha[i];

    // P: C layout -> LDS -> A layout (in-wave transpose through LDS)
#pragma unroll
    for (int g = 0; g < 2; ++g)
#pragma unroll
      for (int i = 0; i < 8; ++i)
        sP[wave][i + 8 * halfl][16 * g + l16] = f2bf(s[g][i]);
    asm volatile("s_wait_dscnt 0" ::: "memory");

    Frag16 pf;
    pf.q[0] = *(const uint4*)&sP[wave][l16][halfl * 8];
    pf.q[1] = *(const uint4*)&sP[wave][l16][halfl * 8 + 16];
#pragma unroll
    for (int fn = 0; fn < 4; ++fn) { // O += P V : n = hd(16fn+l16), kk = key halfl*8..+7, +16
      Frag16 f;
      f.q[0] = *(const uint4*)&sVt[16 * fn + l16][halfl * 8];
      f.q[1] = *(const uint4*)&sVt[16 * fn + l16][halfl * 8 + 16];
      o[fn] = __builtin_amdgcn_wmma_f32_16x16x32_bf16(false, pf.v, false, f.v,
                                                      (short)0, o[fn], false, false);
    }
    __syncthreads();
  }

  // finalize and write [BATCH, SEQ, NH*DHEAD]
#pragma unroll
  for (int i = 0; i < 8; ++i) lrow[i] = 1.f / lrow[i];
#pragma unroll
  for (int fn = 0; fn < 4; ++fn)
#pragma unroll
    for (int i = 0; i < 8; ++i) {
      int ql = qt * 64 + wave * 16 + 8 * halfl + i;
      Oout[((size_t)b * SEQ + ql) * DMODEL + h * DHEAD + 16 * fn + l16] =
          f2bf(o[fn][i] * lrow[i]);
    }
}

// ---------------------------------------------------------------- launch
extern "C" void kernel_launch(void* const* d_in, const int* in_sizes, int n_in,
                              void* d_out, int out_size, void* d_ws, size_t ws_size,
                              hipStream_t stream) {
  (void)in_sizes; (void)n_in; (void)out_size; (void)ws_size;
  const float* x  = (const float*)d_in[0];
  const float* Wq = (const float*)d_in[1];
  const float* Wk = (const float*)d_in[2];
  const float* Wv = (const float*)d_in[3];
  const float* Wo = (const float*)d_in[4];
  // d_in[5] = mask: causal mask computed analytically in k_flash.

  const size_t ML = (size_t)BATCH * SEQ;
  const size_t nX  = ML * DMODEL;
  const size_t nWq = (size_t)DMODEL * DMODEL;
  const size_t nWk = (size_t)(NKV * DHEAD) * DMODEL;

  char* ws = (char*)d_ws;
  size_t off = 0;
  auto carve = [&](size_t bytes) { char* p = ws + off; off = (off + bytes + 255) & ~(size_t)255; return p; };
  __bf16* xb   = (__bf16*)carve(nX * 2);
  __bf16* wqb  = (__bf16*)carve(nWq * 2);
  __bf16* wkb  = (__bf16*)carve(nWk * 2);
  __bf16* wvb  = (__bf16*)carve(nWk * 2);
  __bf16* wob  = (__bf16*)carve(nWq * 2);
  __bf16* qproj= (__bf16*)carve(nX * 2); // reused as attn_out later
  __bf16* kproj= (__bf16*)carve(ML * (NKV * DHEAD) * 2);
  __bf16* vproj= (__bf16*)carve(ML * (NKV * DHEAD) * 2);
  __bf16* qT   = (__bf16*)carve(nX * 2);
  __bf16* kT   = (__bf16*)carve(ML * (NKV * DHEAD) * 2);
  __bf16* vT   = (__bf16*)carve(ML * (NKV * DHEAD) * 2);
  float*  cosT = (float*)carve((size_t)SEQ * DHEAD * 4);
  float*  sinT = (float*)carve((size_t)SEQ * DHEAD * 4);
  __bf16* attn = qproj; // alias: qproj fully consumed before k_flash writes

  auto cvt = [&](const float* src, __bf16* dst, size_t n) {
    int blocks = (int)((n + 1023) / 1024);
    if (blocks > 4096) blocks = 4096;
    k_cvt<<<blocks, 1024, 0, stream>>>(src, dst, (int)n);
  };
  cvt(x, xb, nX);
  cvt(Wq, wqb, nWq);
  cvt(Wk, wkb, nWk);
  cvt(Wv, wvb, nWk);
  cvt(Wo, wob, nWq);

  k_rope_tab<<<(SEQ * DHEAD + 255) / 256, 256, 0, stream>>>(cosT, sinT);

  k_gemm<false><<<dim3(DMODEL / 64, (int)(ML / 128)), 256, 0, stream>>>(
      xb, wqb, qproj, (int)ML, DMODEL, DMODEL);
  k_gemm<false><<<dim3((NKV * DHEAD) / 64, (int)(ML / 128)), 256, 0, stream>>>(
      xb, wkb, kproj, (int)ML, NKV * DHEAD, DMODEL);
  k_gemm<false><<<dim3((NKV * DHEAD) / 64, (int)(ML / 128)), 256, 0, stream>>>(
      xb, wvb, vproj, (int)ML, NKV * DHEAD, DMODEL);

  {
    int tq = BATCH * SEQ * NH * DHEAD, tk = BATCH * SEQ * NKV * DHEAD;
    k_rope_perm<<<(tq + 1023) / 1024, 1024, 0, stream>>>(qproj, qT, cosT, sinT, NH, 1);
    k_rope_perm<<<(tk + 1023) / 1024, 1024, 0, stream>>>(kproj, kT, cosT, sinT, NKV, 1);
    k_rope_perm<<<(tk + 1023) / 1024, 1024, 0, stream>>>(vproj, vT, cosT, sinT, NKV, 0);
  }

  k_flash<<<dim3(SEQ / 64, NH, BATCH), 128, 0, stream>>>(qT, kT, vT, attn);

  k_gemm<true><<<dim3(DMODEL / 64, (int)(ML / 128)), 256, 0, stream>>>(
      attn, wob, d_out, (int)ML, DMODEL, DMODEL);
}
